// SwinTransformerLayer_9921374454129
// MI455X (gfx1250) — matrix operand order repaired and checked by reading, not compiled, not run on hardware
//
#include <hip/hip_runtime.h>
#include <hip/hip_bf16.h>

// ---------------------------------------------------------------------------
// Swin Transformer layer for MI455X (gfx1250), wave32 + WMMA f16->f32.
// All GEMMs use v_wmma_f32_16x16x32_f16. B operands are kept N-major
// ("transposed") in LDS so every B fragment is one aligned 32B LDS load;
// A fragments are one aligned 32B LDS load as well. Contiguous LDS tiles are staged
// with the Tensor Data Mover (tensor_load_to_lds + s_wait_tensorcnt) when the
// builtin is available; an opaque-index dummy store keeps the compiler from
// folding the (TDM-written) LDS reads to undef. Fallback: uint4 b128 copies.
// 128-thread blocks = 4 wave32 waves; each wave owns a 16-row stripe.
// ---------------------------------------------------------------------------

typedef __attribute__((ext_vector_type(16))) _Float16 v16h;
typedef __attribute__((ext_vector_type(8)))  _Float16 v8h;
typedef __attribute__((ext_vector_type(8)))  float    v8f;
typedef __attribute__((ext_vector_type(4)))  unsigned int u32x4;
typedef __attribute__((ext_vector_type(8)))  int          i32x8;
typedef __attribute__((ext_vector_type(4)))  int          i32x4;
typedef _Float16 h16;
typedef unsigned int u32;

#define WSZ     7
#define SHIFT_  3
#define NTOK    49        // WSZ*WSZ
#define NP      64        // padded tokens per window (4 WMMA M-tiles)
#define CC      96        // channels
#define HD      32        // head dim
#define NHEAD   3
#define HH      112
#define WWID    112
#define NWIN    4096      // 16 * (112/7)^2
#define NTOKTOT 200704    // 16*112*112
#define MTILES  3136      // NTOKTOT / NP (exact)

// ---------------- workspace layout (bytes), phase-aliased ----------------
#define OFF_WINS   ((size_t)0)
#define OFF_QKV    ((size_t)50331648)     // 48 MB
#define OFF_ATTNW  ((size_t)0)
#define OFF_H1     ((size_t)0)            // 154,140,672
#define OFF_XN     ((size_t)154140672)    //  38,535,168
#define OFF_OUT1   ((size_t)201326592)    // 192 MB
#define OFF_WQ     ((size_t)278396928)    // transposed f16 weights follow
#define OFF_WP     ((size_t)278452224)
#define OFF_W1     ((size_t)278470656)
#define OFF_W2     ((size_t)278544384)

// ---------------- TDM availability ----------------
#if defined(__AMDGCN__) && __has_builtin(__builtin_amdgcn_tensor_load_to_lds)
#define USE_TDM 1
#else
#define USE_TDM 0
#endif

#if USE_TDM
// 1-D contiguous DMA: global -> LDS, nbytes % 8 == 0, both 8B-aligned.
// D# per CDNA5 ISA:
//  group0: d0 = count(1); d1 = lds byte addr; d2/d3 = 57b global addr, type=2
//  group1: data_size=8B, tensor_dim0 = tile_dim0 = ne, tensor_dim1 = 1,
//          dim0_stride = ne, everything else 0.
__device__ __forceinline__ void tdm_load_1d(unsigned lds_off, const void* g,
                                            unsigned nbytes) {
  unsigned long long ga = (unsigned long long)(size_t)g;
  unsigned ne = nbytes >> 3;                       // 8-byte elements (<= 65535)
  u32x4 g0;
  g0[0] = 1u;                                      // count = 1 valid descriptor
  g0[1] = lds_off;                                 // lds_addr
  g0[2] = (unsigned)ga;                            // global_addr[31:0]
  g0[3] = (unsigned)((ga >> 32) & 0x01ffffffu) | 0x80000000u;  // [56:32] | type=2
  i32x8 g1;
  g1[0] = (int)(3u << 16);                         // data_size = 3 (8 bytes)
  g1[1] = (int)((ne & 0xffffu) << 16);             // tensor_dim0[15:0]
  g1[2] = (int)(((ne >> 16) & 0xffffu) | (1u << 16)); // dim0 hi | tensor_dim1=1
  g1[3] = (int)(ne << 16);                         // tile_dim0
  g1[4] = 1;                                       // tile_dim1 = 1
  g1[5] = (int)ne;                                 // tensor_dim0_stride lo
  g1[6] = 0;
  g1[7] = 0;
  i32x4 z4; z4[0] = 0; z4[1] = 0; z4[2] = 0; z4[3] = 0;
#if defined(__clang_major__) && (__clang_major__ >= 23)
  i32x8 z8;
  z8[0] = 0; z8[1] = 0; z8[2] = 0; z8[3] = 0;
  z8[4] = 0; z8[5] = 0; z8[6] = 0; z8[7] = 0;
  __builtin_amdgcn_tensor_load_to_lds(g0, g1, z4, z4, z8, 0);   // 6-arg form
#else
  __builtin_amdgcn_tensor_load_to_lds(g0, g1, z4, z4, 0);       // 5-arg form
#endif
}

// value the compiler cannot reason about (but is `v` at runtime)
__device__ __forceinline__ int opaque_i(int v) {
  asm volatile("" : "+v"(v));
  return v;
}
#endif

// ---------------- WMMA helpers (ISA 7.12.2 layouts) ----------------

__device__ __forceinline__ v8f wmma_f16(v16h a, v16h b, v8f c) {
  return __builtin_amdgcn_wmma_f32_16x16x32_f16(
      false, a, false, b, (short)0, c, false, false);
}

// A fragment: 16x32 f16 tile from row-major source (row stride ld halves,
// k offset k0; ld % 16 == 0, k0 % 32 == 0 -> one 32B-aligned v16h load, k = 16h + s).
__device__ __forceinline__ v16h load_a_frag(const h16* p, int ld, int k0) {
  int lane = threadIdx.x & 31;
  const h16* q = p + (lane & 15) * ld + k0 + ((lane >> 4) << 4);
  return *(const v16h*)(q);
}

// B fragment from N-major source: element (k,n) at p[n*ld + k].
// Requires ld % 16 == 0 and k0 % 32 == 0 -> one 32B v16h load.
__device__ __forceinline__ v16h load_bt_frag(const h16* p, int ld, int k0) {
  int lane = threadIdx.x & 31;
  int col  = lane & 15;
  int kb   = (lane >> 4) << 4;
  return *(const v16h*)(p + col * ld + k0 + kb);
}

// uint4 (b128) cooperative copy, nhalves % 8 == 0, 16B-aligned both sides.
__device__ __forceinline__ void copy16(h16* dst, const h16* src, int nhalves,
                                       int t, int nt) {
  uint4* d = (uint4*)dst;
  const uint4* s = (const uint4*)src;
  int n = nhalves >> 3;
  for (int i = t; i < n; i += nt) d[i] = s[i];
}

// Stage A (at smem[0..na)) and B (at smem[na..na+nb)) tiles.
// smem must have >= na+nb+8 elements; the last slot is a dummy written through
// an opaque index so the compiler cannot treat TDM-filled LDS as undef.
__device__ __forceinline__ void stage2(h16* smem, int na, int nb,
                                       const h16* asrc, const h16* bsrc, int t) {
#if USE_TDM
  if (t < 32) {                       // one wave issues both DMAs
    tdm_load_1d(0u, asrc, (unsigned)na * 2u);
    tdm_load_1d((unsigned)na * 2u, bsrc, (unsigned)nb * 2u);
    __builtin_amdgcn_s_wait_tensorcnt(0);
  }
  if (t == 0) smem[opaque_i(na + nb)] = (h16)0.f;   // pad slot, opaque index
#else
  copy16(smem,      asrc, na, t, 128);
  copy16(smem + na, bsrc, nb, t, 128);
#endif
  __syncthreads();
}

__device__ __forceinline__ u32 pack2(float a, float b) {
  union { h16 h; unsigned short u; } c0, c1;
  c0.h = (h16)a; c1.h = (h16)b;
  return (u32)c0.u | ((u32)c1.u << 16);
}

// ---------------- k0: weight f32 -> f16, transposed to N-major ----------------
__global__ void k0_cvt_t(const float* __restrict__ src, h16* __restrict__ dst,
                         int K, int N) {
  int i = blockIdx.x * blockDim.x + threadIdx.x;
  if (i >= K * N) return;
  int n = i / K, k = i - n * K;
  dst[i] = (h16)src[k * N + n];
}

// ---------------- k1: LN1 + cyclic shift + window partition ----------------
__global__ void k1_ln1_part(const float* __restrict__ x,
                            const float* __restrict__ g,
                            const float* __restrict__ bta,
                            h16* __restrict__ wins) {
  int idx = blockIdx.x * blockDim.x + threadIdx.x;   // NWIN*NP tokens
  if (idx >= NWIN * NP) return;
  int w = idx >> 6, n = idx & 63;
  u32* dst = (u32*)(wins + (size_t)idx * CC);
  if (n >= NTOK) {
    for (int c = 0; c < CC / 2; ++c) dst[c] = 0u;
    return;
  }
  int bi = w >> 8, wi = w & 255;
  int wh = wi >> 4, wwi = wi & 15;
  int r = n / WSZ, cc = n % WSZ;
  int h  = (wh * WSZ + r  + SHIFT_) % HH;
  int wc = (wwi * WSZ + cc + SHIFT_) % WWID;
  const float* row = x + (((size_t)bi * HH + h) * WWID + wc) * CC;
  float mu = 0.f;
  for (int c = 0; c < CC; ++c) mu += row[c];
  mu *= (1.f / CC);
  float var = 0.f;
  for (int c = 0; c < CC; ++c) { float d = row[c] - mu; var += d * d; }
  var *= (1.f / CC);
  float inv = rsqrtf(var + 1e-4f);
  for (int c = 0; c < CC; c += 2) {
    float a0 = (row[c]     - mu) * inv * g[c]     + bta[c];
    float a1 = (row[c + 1] - mu) * inv * g[c + 1] + bta[c + 1];
    dst[c >> 1] = pack2(a0, a1);
  }
}

// ---------------- k2: QKV GEMM (64x96 per block, K=96) ----------------
__global__ void k2_qkv(const h16* __restrict__ wins, const h16* __restrict__ wqT,
                       const float* __restrict__ qkv_b, h16* __restrict__ qkv) {
  __shared__ alignas(128) h16 smem[NP * CC + CC * CC + 8];
  h16* As  = smem;              // 64x96 activations, row-major
  h16* BsT = smem + NP * CC;    // 96(N) x 96(K), N-major
  int w = blockIdx.x, slice = blockIdx.y;
  int t = threadIdx.x;
  stage2(smem, NP * CC, CC * CC,
         wins + (size_t)w * NP * CC, wqT + (size_t)slice * CC * CC, t);
  int wave = t >> 5, lane = t & 31, mrow = wave * 16;
  v16h a[3];
#pragma unroll
  for (int kk = 0; kk < 3; ++kk) a[kk] = load_a_frag(As + mrow * CC, CC, kk * 32);
  float scl = (slice == 0) ? 0.17677669529663687f : 1.f;   // hd^-0.5 on Q
#pragma unroll
  for (int nt = 0; nt < 6; ++nt) {
    v8f acc = {};
#pragma unroll
    for (int kk = 0; kk < 3; ++kk)
      acc = wmma_f16(a[kk], load_bt_frag(BsT + nt * 16 * CC, CC, kk * 32), acc);
    int col  = nt * 16 + (lane & 15);
    int head = col / HD, d = col % HD;
    float bias = qkv_b[slice * CC + col];
    int baseRow = mrow + ((lane >> 4) << 3);
    h16* dstb = qkv + (((size_t)w * 3 + slice) * NHEAD + head) * NP * HD + d;
#pragma unroll
    for (int r = 0; r < 8; ++r)
      dstb[(baseRow + r) * HD] = (h16)((acc[r] + bias) * scl);
  }
}

// ---------------- k3: attention per (window, head) ----------------
__global__ void k3_attn(const h16* __restrict__ qkv,
                        const float* __restrict__ rel_bias,
                        h16* __restrict__ attnw) {
  __shared__ alignas(32) h16   Qs [NP * HD];   // token-major (A operand)
  __shared__ alignas(32) h16   Ks [NP * HD];   // token-major == K^T N-major
  __shared__ alignas(32) h16   VsT[HD * NP];   // d-major (B operand for P*V)
  __shared__ alignas(32) float Sf [NP * NP];
  __shared__ alignas(32) h16   Ph [NP * NP];
  int w = blockIdx.x / NHEAD, head = blockIdx.x % NHEAD;
  int t = threadIdx.x;
  const h16* base = qkv + (size_t)w * 3 * NHEAD * NP * HD;
  const h16* qsrc = base + (0 * NHEAD + head) * NP * HD;
  const h16* ksrc = base + (1 * NHEAD + head) * NP * HD;
  const h16* vsrc = base + (2 * NHEAD + head) * NP * HD;
  copy16(Qs, qsrc, NP * HD, t, 128);
  copy16(Ks, ksrc, NP * HD, t, 128);
  for (int i = t; i < NP * HD; i += 128) {       // transpose V into VsT
    int tok = i >> 5, d = i & 31;
    VsT[d * NP + tok] = vsrc[i];
  }
  __syncthreads();

  int wave = t >> 5, lane = t & 31, mrow = wave * 16;
  int wi = w & 255, wh = wi >> 4, wwi = wi & 15;

  // S = Q * K^T : Ks is already N-major for the B operand (ld = HD).
  v16h aq = load_a_frag(Qs + mrow * HD, HD, 0);
#pragma unroll
  for (int nt = 0; nt < 4; ++nt) {
    v8f acc = {};
    acc = wmma_f16(aq, load_bt_frag(Ks + nt * 16 * HD, HD, 0), acc);
    int ncol = nt * 16 + (lane & 15);
    int baseRow = mrow + ((lane >> 4) << 3);
#pragma unroll
    for (int r = 0; r < 8; ++r) {
      int m = baseRow + r;
      float sv;
      if (ncol >= NTOK)      sv = -1e30f;     // padded key columns
      else if (m >= NTOK)    sv = 0.f;        // padded query rows: harmless
      else {
        int r1 = m / WSZ, c1 = m % WSZ, r2 = ncol / WSZ, c2 = ncol % WSZ;
        int h1 = wh * WSZ + r1, w1 = wwi * WSZ + c1;
        int h2 = wh * WSZ + r2, w2 = wwi * WSZ + c2;
        int g1 = (h1 < 105 ? 0 : (h1 < 109 ? 1 : 2)) * 3 + (w1 < 105 ? 0 : (w1 < 109 ? 1 : 2));
        int g2 = (h2 < 105 ? 0 : (h2 < 109 ? 1 : 2)) * 3 + (w2 < 105 ? 0 : (w2 < 109 ? 1 : 2));
        float msk = (g1 != g2) ? -100.f : 0.f;
        int ridx = (r1 - r2 + WSZ - 1) * (2 * WSZ - 1) + (c1 - c2 + WSZ - 1);
        sv = acc[r] + rel_bias[ridx * NHEAD + head] + msk;
      }
      Sf[m * NP + ncol] = sv;
    }
  }
  __syncthreads();

  // row softmax, one row per thread; write P as packed 32-bit pairs
  if (t < NP) {
    float* row = Sf + t * NP;
    float mx = -1e30f;
    for (int j = 0; j < NP; ++j) mx = fmaxf(mx, row[j]);
    float s = 0.f;
    for (int j = 0; j < NP; ++j) s += __expf(row[j] - mx);
    float inv = 1.f / s;
    u32* prow = (u32*)(Ph + t * NP);
    for (int j = 0; j < NP / 2; ++j)
      prow[j] = pack2(__expf(row[2 * j] - mx) * inv,
                      __expf(row[2 * j + 1] - mx) * inv);
  }
  __syncthreads();

  // O = P * V  (64x32, K=64): B from VsT (N-major over d, ld = NP)
#pragma unroll
  for (int nt = 0; nt < 2; ++nt) {
    v8f acc = {};
#pragma unroll
    for (int kk = 0; kk < 2; ++kk) {
      v16h ap = load_a_frag(Ph + mrow * NP, NP, kk * 32);
      acc = wmma_f16(ap, load_bt_frag(VsT + nt * 16 * NP, NP, kk * 32), acc);
    }
    int col = head * HD + nt * 16 + (lane & 15);
    int baseRow = mrow + ((lane >> 4) << 3);
    h16* dst = attnw + (size_t)w * NP * CC + col;
#pragma unroll
    for (int r = 0; r < 8; ++r) dst[(baseRow + r) * CC] = (h16)acc[r];
  }
}

// ---------------- k4: proj GEMM + window reverse + residual ----------------
__global__ void k4_proj(const h16* __restrict__ attnw, const h16* __restrict__ wpT,
                        const float* __restrict__ proj_b,
                        const float* __restrict__ x, float* __restrict__ out1) {
  __shared__ alignas(128) h16 smem[NP * CC + CC * CC + 8];
  h16* As  = smem;
  h16* BsT = smem + NP * CC;
  int w = blockIdx.x, t = threadIdx.x;
  stage2(smem, NP * CC, CC * CC, attnw + (size_t)w * NP * CC, wpT, t);
  int wave = t >> 5, lane = t & 31, mrow = wave * 16;
  int bi = w >> 8, wi = w & 255, wh = wi >> 4, wwi = wi & 15;
  v16h a[3];
#pragma unroll
  for (int kk = 0; kk < 3; ++kk) a[kk] = load_a_frag(As + mrow * CC, CC, kk * 32);
#pragma unroll
  for (int nt = 0; nt < 6; ++nt) {
    v8f acc = {};
#pragma unroll
    for (int kk = 0; kk < 3; ++kk)
      acc = wmma_f16(a[kk], load_bt_frag(BsT + nt * 16 * CC, CC, kk * 32), acc);
    int col = nt * 16 + (lane & 15);
    float pb = proj_b[col];
    int baseRow = mrow + ((lane >> 4) << 3);
#pragma unroll
    for (int r = 0; r < 8; ++r) {
      int n = baseRow + r;
      if (n < NTOK) {
        int r1 = n / WSZ, c1 = n % WSZ;
        int h  = (wh * WSZ + r1 + SHIFT_) % HH;     // reverse shift
        int wc = (wwi * WSZ + c1 + SHIFT_) % WWID;
        size_t o = (((size_t)bi * HH + h) * WWID + wc) * CC + col;
        out1[o] = x[o] + acc[r] + pb;
      }
    }
  }
}

// ---------------- k5a: LN2 ----------------
__global__ void k5a_ln2(const float* __restrict__ out1, const float* __restrict__ g,
                        const float* __restrict__ bta, h16* __restrict__ xn) {
  int tk = blockIdx.x * blockDim.x + threadIdx.x;
  if (tk >= NTOKTOT) return;
  const float* row = out1 + (size_t)tk * CC;
  float mu = 0.f;
  for (int c = 0; c < CC; ++c) mu += row[c];
  mu *= (1.f / CC);
  float var = 0.f;
  for (int c = 0; c < CC; ++c) { float d = row[c] - mu; var += d * d; }
  var *= (1.f / CC);
  float inv = rsqrtf(var + 1e-4f);
  u32* dst = (u32*)(xn + (size_t)tk * CC);
  for (int c = 0; c < CC; c += 2) {
    float a0 = (row[c]     - mu) * inv * g[c]     + bta[c];
    float a1 = (row[c + 1] - mu) * inv * g[c + 1] + bta[c + 1];
    dst[c >> 1] = pack2(a0, a1);
  }
}

// ---------------- k5b: FC1 GEMM + GELU (64x128 tile, K=96) ----------------
__global__ void k5b_fc1(const h16* __restrict__ xn, const h16* __restrict__ w1T,
                        const float* __restrict__ b1, h16* __restrict__ h1out) {
  __shared__ alignas(128) h16 smem[NP * CC + 128 * CC + 8];
  h16* As  = smem;              // 64x96
  h16* BsT = smem + NP * CC;    // 128(N) x 96(K)
  int mb = blockIdx.x, slice = blockIdx.y;
  int t = threadIdx.x;
  stage2(smem, NP * CC, 128 * CC,
         xn + (size_t)mb * NP * CC, w1T + (size_t)slice * 128 * CC, t);
  int wave = t >> 5, lane = t & 31, mrow = wave * 16;
  v16h a[3];
#pragma unroll
  for (int kk = 0; kk < 3; ++kk) a[kk] = load_a_frag(As + mrow * CC, CC, kk * 32);
#pragma unroll
  for (int nt = 0; nt < 8; ++nt) {
    v8f acc = {};
#pragma unroll
    for (int kk = 0; kk < 3; ++kk)
      acc = wmma_f16(a[kk], load_bt_frag(BsT + nt * 16 * CC, CC, kk * 32), acc);
    int col = slice * 128 + nt * 16 + (lane & 15);
    float bb = b1[col];
    int baseRow = mb * NP + mrow + ((lane >> 4) << 3);
#pragma unroll
    for (int r = 0; r < 8; ++r) {
      float v = acc[r] + bb;
      float gel = 0.5f * v * (1.f + tanhf(0.7978845608028654f * (v + 0.044715f * v * v * v)));
      h1out[(size_t)(baseRow + r) * 384 + col] = (h16)gel;
    }
  }
}

// ---------------- k6: FC2 GEMM (K=384 chunked) + residual ----------------
__global__ void k6_fc2(const h16* __restrict__ h1, const h16* __restrict__ w2T,
                       const float* __restrict__ b2, const float* __restrict__ out1,
                       float* __restrict__ out) {
  __shared__ alignas(32) h16 As [NP * CC];    // 64 x 96 K-chunk
  __shared__ alignas(32) h16 BsT[CC * CC];    // 96(N) x 96(K) K-chunk
  int mb = blockIdx.x, t = threadIdx.x;
  int wave = t >> 5, lane = t & 31, mrow = wave * 16;
  v8f acc[6];
  v8f zero = {};
#pragma unroll
  for (int i = 0; i < 6; ++i) acc[i] = zero;
  for (int kc = 0; kc < 4; ++kc) {
    __syncthreads();
    {
      uint4* d = (uint4*)As;
      for (int i = t; i < NP * 12; i += 128) {
        int rr = i / 12, j = i - rr * 12;
        const uint4* s = (const uint4*)(h1 + (size_t)(mb * NP + rr) * 384 + kc * CC);
        d[i] = s[j];
      }
      uint4* db = (uint4*)BsT;
      for (int i = t; i < CC * 12; i += 128) {
        int n = i / 12, j = i - n * 12;
        const uint4* s = (const uint4*)(w2T + (size_t)n * 384 + kc * CC);
        db[i] = s[j];
      }
      if (kc < 3 && t < NP)   // prefetch next A chunk (global_prefetch_b8)
        __builtin_prefetch(h1 + (size_t)(mb * NP + t) * 384 + (kc + 1) * CC, 0, 0);
    }
    __syncthreads();
    v16h a[3];
#pragma unroll
    for (int kk = 0; kk < 3; ++kk) a[kk] = load_a_frag(As + mrow * CC, CC, kk * 32);
#pragma unroll
    for (int nt = 0; nt < 6; ++nt)
#pragma unroll
      for (int kk = 0; kk < 3; ++kk)
        acc[nt] = wmma_f16(a[kk], load_bt_frag(BsT + nt * 16 * CC, CC, kk * 32), acc[nt]);
  }
#pragma unroll
  for (int nt = 0; nt < 6; ++nt) {
    int col = nt * 16 + (lane & 15);
    float bb = b2[col];
    int baseRow = mb * NP + mrow + ((lane >> 4) << 3);
#pragma unroll
    for (int r = 0; r < 8; ++r) {
      size_t o = (size_t)(baseRow + r) * CC + col;
      out[o] = out1[o] + acc[nt][r] + bb;
    }
  }
}

// ---------------------------------------------------------------------------
extern "C" void kernel_launch(void* const* d_in, const int* in_sizes, int n_in,
                              void* d_out, int out_size, void* d_ws, size_t ws_size,
                              hipStream_t stream) {
  (void)in_sizes; (void)n_in; (void)out_size; (void)ws_size;
  const float* x      = (const float*)d_in[0];
  const float* qkv_w  = (const float*)d_in[1];
  const float* qkv_b  = (const float*)d_in[2];
  const float* proj_w = (const float*)d_in[3];
  const float* proj_b = (const float*)d_in[4];
  const float* relb   = (const float*)d_in[5];
  const float* ln1g   = (const float*)d_in[6];
  const float* ln1b   = (const float*)d_in[7];
  const float* ln2g   = (const float*)d_in[8];
  const float* ln2b   = (const float*)d_in[9];
  const float* fc1w   = (const float*)d_in[10];
  const float* fc1b   = (const float*)d_in[11];
  const float* fc2w   = (const float*)d_in[12];
  const float* fc2b   = (const float*)d_in[13];
  float* out = (float*)d_out;

  char* ws = (char*)d_ws;
  h16*   wins  = (h16*)(ws + OFF_WINS);
  h16*   qkvh  = (h16*)(ws + OFF_QKV);
  h16*   attnw = (h16*)(ws + OFF_ATTNW);
  h16*   h1b   = (h16*)(ws + OFF_H1);
  h16*   xnh   = (h16*)(ws + OFF_XN);
  float* out1  = (float*)(ws + OFF_OUT1);
  h16*   wqT   = (h16*)(ws + OFF_WQ);    // [288][96]
  h16*   wpT   = (h16*)(ws + OFF_WP);    // [96][96]
  h16*   w1T   = (h16*)(ws + OFF_W1);    // [384][96]
  h16*   w2T   = (h16*)(ws + OFF_W2);    // [96][384]

  // weights -> f16, transposed to N-major (cheap, deterministic each call)
  k0_cvt_t<<<(27648 + 255) / 256, 256, 0, stream>>>(qkv_w, wqT, 96, 288);
  k0_cvt_t<<<(9216  + 255) / 256, 256, 0, stream>>>(proj_w, wpT, 96, 96);
  k0_cvt_t<<<(36864 + 255) / 256, 256, 0, stream>>>(fc1w, w1T, 96, 384);
  k0_cvt_t<<<(36864 + 255) / 256, 256, 0, stream>>>(fc2w, w2T, 384, 96);

  k1_ln1_part<<<(NWIN * NP) / 256, 256, 0, stream>>>(x, ln1g, ln1b, wins);
  k2_qkv<<<dim3(NWIN, 3), 128, 0, stream>>>(wins, wqT, qkv_b, qkvh);
  k3_attn<<<NWIN * NHEAD, 128, 0, stream>>>(qkvh, relb, attnw);
  k4_proj<<<NWIN, 128, 0, stream>>>(attnw, wpT, proj_b, x, out1);
  k5a_ln2<<<(NTOKTOT + 255) / 256, 256, 0, stream>>>(out1, ln2g, ln2b, xnh);
  k5b_fc1<<<dim3(MTILES, 3), 128, 0, stream>>>(xnh, w1T, fc1b, h1b);
  k6_fc2<<<MTILES, 128, 0, stream>>>(h1b, w2T, fc2b, out1, out);
}
